// EMA_26456998543834
// MI455X (gfx1250) — matrix operand order repaired and checked by reading, not compiled, not run on hardware
//
#include <hip/hip_runtime.h>
#include <cstdint>
#include <cstddef>

typedef __attribute__((ext_vector_type(16))) _Float16 v16h;
typedef __attribute__((ext_vector_type(8)))  _Float16 v8h;
typedef __attribute__((ext_vector_type(8)))  float    v8f;
typedef __attribute__((ext_vector_type(4)))  unsigned int u32x4;
typedef __attribute__((ext_vector_type(8)))  int          i32x8;
typedef __attribute__((ext_vector_type(4)))  int          i32x4;

#define CC    512
#define HH    160
#define WW_   160
#define HWSZ  25600
#define SLEN  320
#define KTOT  4608   /* 9 * 512 */
#define HP    162    /* padded height/width */
#define PHW   26244  /* 162*162 */

__device__ __forceinline__ float sigmoidf(float x){ return 1.0f/(1.0f+__expf(-x)); }

__device__ __forceinline__ float block_reduce_sum_256(float v, float* sb){
  int t = threadIdx.x;
  sb[t] = v; __syncthreads();
  #pragma unroll
  for (int s=128; s>0; s>>=1){ if (t<s) sb[t]+=sb[t+s]; __syncthreads(); }
  float r = sb[0]; __syncthreads();
  return r;
}

// Async global->LDS copy of 16 bytes per lane. LDS offset = low 32 bits of the
// generic pointer (flat LDS aperture keeps the wave-relative offset in addr[31:0]).
__device__ __forceinline__ void async_copy_b128(const void* gp, void* lp){
  unsigned lo = (unsigned)(uintptr_t)lp;
  unsigned long long ga = (unsigned long long)(uintptr_t)gp;
  asm volatile("global_load_async_to_lds_b128 %0, %1, off" :: "v"(lo), "v"(ga) : "memory");
}
__device__ __forceinline__ void wait_async0(){
  asm volatile("s_wait_asynccnt 0x0" ::: "memory");
}

// TDM: DMA a 2D tile (tile_dim1 rows x tile_dim0 elements of 2B) from global to LDS
// with 16B padding inserted after every 64B row (LDS row stride 80B). D# packing per
// CDNA5 ISA sec.8. Issued by one wave (EXEC is ignored by TDM).
// This toolchain's builtin takes 6 args: (g0 v4u, g1 v8i, g2 v4i, g3 v4i, v8i, cpol).
__device__ __forceinline__ void tdm_load_weights(const _Float16* gsrc, void* ldst){
  unsigned long long ga = (unsigned long long)(uintptr_t)(const void*)gsrc;
  unsigned ldsoff = (unsigned)(uintptr_t)ldst;
  u32x4 g0;
  g0[0] = 1u;                                   // count=1 (valid user descriptor)
  g0[1] = ldsoff;                               // lds_addr (bytes)
  g0[2] = (unsigned)ga;                         // global_addr[31:0]
  g0[3] = (unsigned)((ga >> 32) & 0x01FFFFFFu)  // global_addr[56:32]
        | (2u << 30);                           // type = 2 ("image")
  i32x8 g1;
  g1[0] = (1<<16)                               // data_size = 2 bytes
        | (1<<20)                               // pad_enable
        | (3<<22)                               // pad_interval: 16 DWORDs (64B)
        | (3<<25);                              // pad_amount:   4 DWORDs (16B)
  g1[1] = (int)(32u  << 16);                    // tensor_dim0 = 32 (tile-relative)
  g1[2] = (int)(128u << 16);                    // tensor_dim1 = 128
  g1[3] = (int)(32u  << 16);                    // tile_dim0 = 32 elements (64B row)
  g1[4] = 128;                                  // tile_dim1 = 128 rows, tile_dim2 = 0
  g1[5] = KTOT;                                 // tensor_dim0_stride = 4608 elements
  g1[6] = 0;
  g1[7] = 0;
  i32x4 gz4 = {0,0,0,0};                        // groups 2/3 unused (2D tensor)
  i32x8 gz8 = {0,0,0,0,0,0,0,0};
  __builtin_amdgcn_tensor_load_to_lds(g0, g1, gz4, gz4, gz8, 0);
}

// ---------------- 1) H/W mean pools -> cat (512 x 320) ----------------
__global__ __launch_bounds__(256) void pool_kernel(const float* __restrict__ x,
                                                   float* __restrict__ cat){
  __shared__ float rs[HH];
  __shared__ float cs[WW_];
  int c = blockIdx.x, t = threadIdx.x;
  if (t < HH){ rs[t]=0.f; cs[t]=0.f; }
  __syncthreads();
  const float* xc = x + (size_t)c*HWSZ;
  for (int i=t; i<HWSZ; i+=256){
    float v = xc[i];
    atomicAdd(&rs[i/WW_], v);
    atomicAdd(&cs[i%WW_], v);
  }
  __syncthreads();
  if (t < HH){
    cat[(size_t)c*SLEN + t]      = rs[t]*(1.0f/WW_);
    cat[(size_t)c*SLEN + HH + t] = cs[t]*(1.0f/HH);
  }
}

// ---------------- 2) hw_feat = sigmoid(w1 @ cat + b1) -> sig (512 x 320) ----------------
__global__ __launch_bounds__(256) void hwfeat_kernel(const float* __restrict__ w1,
                                                     const float* __restrict__ b1,
                                                     const float* __restrict__ cat,
                                                     float* __restrict__ sig){
  int o = blockIdx.x, t = threadIdx.x;
  const float* wr = w1 + (size_t)o*CC;
  float acc0 = 0.f, acc1 = 0.f;
  for (int c=0; c<CC; ++c){
    float wv = wr[c];
    acc0 += wv * cat[(size_t)c*SLEN + t];
    if (t < SLEN-256) acc1 += wv * cat[(size_t)c*SLEN + t + 256];
  }
  float bb = b1[o];
  sig[(size_t)o*SLEN + t] = sigmoidf(acc0 + bb);
  if (t < SLEN-256) sig[(size_t)o*SLEN + t + 256] = sigmoidf(acc1 + bb);
}

// ---------------- 3) x2 gate + zero-padded f16 copy of x + per-channel mean ----------------
__global__ __launch_bounds__(256) void x2_kernel(const float* __restrict__ x,
                                                 const float* __restrict__ sig,
                                                 float* __restrict__ x2,
                                                 _Float16* __restrict__ xp,
                                                 float* __restrict__ mean2){
  __shared__ float sb[256];
  int c = blockIdx.x, t = threadIdx.x;
  const float* xc = x + (size_t)c*HWSZ;
  const float* sg = sig + (size_t)c*SLEN;
  _Float16* xpc = xp + (size_t)c*PHW;
  // zero the 1-pixel border of the padded plane
  for (int j=t; j<644; j+=256){
    int hh, ww;
    if (j < 162)      { hh = 0;        ww = j; }
    else if (j < 324) { hh = 161;      ww = j-162; }
    else if (j < 484) { hh = j-324+1;  ww = 0; }
    else              { hh = j-484+1;  ww = 161; }
    xpc[(size_t)hh*HP + ww] = (_Float16)0.f;
  }
  float lsum = 0.f;
  for (int i=t; i<HWSZ; i+=256){
    float xv = xc[i];
    int h = i/WW_, w = i%WW_;
    float v  = xv * sg[h] * sg[HH + w];
    x2[(size_t)c*HWSZ + i] = v;
    xpc[(size_t)(h+1)*HP + (w+1)] = (_Float16)xv;
    lsum += v;
  }
  float s = block_reduce_sum_256(lsum, sb);
  if (t==0) mean2[c] = s * (1.0f/HWSZ);
}

// ---------------- 4) repack w3 (co,ci,ky,kx) fp32 -> wp f16 (co, tap, ci) ----------------
__global__ __launch_bounds__(256) void repack_w3_kernel(const float* __restrict__ w3,
                                                        _Float16* __restrict__ wp){
  int idx = blockIdx.x*256 + threadIdx.x;
  if (idx >= CC*KTOT) return;
  int co  = idx / KTOT;
  int r   = idx % KTOT;
  int tap = r / CC;
  int ci  = r % CC;
  wp[idx] = (_Float16)w3[((size_t)co*CC + ci)*9 + tap];
}

// ---------------- single-block 512-softmax ----------------
__global__ __launch_bounds__(256) void softmax512_kernel(const float* __restrict__ in,
                                                         float* __restrict__ out){
  __shared__ float sb[256];
  int t = threadIdx.x;
  float a = in[t], b = in[t+256];
  sb[t] = fmaxf(a, b); __syncthreads();
  #pragma unroll
  for (int s=128; s>0; s>>=1){ if (t<s) sb[t]=fmaxf(sb[t], sb[t+s]); __syncthreads(); }
  float mx = sb[0]; __syncthreads();
  float ea = __expf(a-mx), eb = __expf(b-mx);
  sb[t] = ea + eb; __syncthreads();
  #pragma unroll
  for (int s=128; s>0; s>>=1){ if (t<s) sb[t]+=sb[t+s]; __syncthreads(); }
  float inv = 1.0f/sb[0];
  out[t]     = ea*inv;
  out[t+256] = eb*inv;
}

// ---------------- 5) 3x3 conv 512->512, implicit GEMM, f16 WMMA ----------------------
// Block: 8 waves, tile 128 out-channels x 32 pixels. K = 9 taps x 512 ci, chunks of 32.
// A (weights) chunk: TDM tensor_load_to_lds with LDS row padding (issued by wave 0).
// B (input) chunk: regular loads + transposed LDS stores (padded input, no branches).
// Double-buffered, one barrier per K-chunk; TDM overlaps the WMMAs.
__device__ __forceinline__ void conv_load_B(int kc, int h, int w0, int t,
                                            const _Float16* __restrict__ xp,
                                            _Float16 (*Bs)[40]){
  int tap   = kc >> 4;
  int cbase = (kc & 15) << 5;
  int ky = tap/3, kx = tap%3;
  int hh = h + ky;            // padded coords (offsets +1 and -1 cancel)
  int wwb = w0 + kx;
  int ci = t >> 3, p0 = (t & 7) * 4;
  const _Float16* src = xp + (size_t)(cbase+ci)*PHW + (size_t)hh*HP + wwb + p0;
  __builtin_prefetch(src + 32, 0, 1);
  _Float16 v0=src[0], v1=src[1], v2=src[2], v3=src[3];
  Bs[p0+0][ci]=v0; Bs[p0+1][ci]=v1; Bs[p0+2][ci]=v2; Bs[p0+3][ci]=v3;
}

__global__ __launch_bounds__(256) void conv_wmma_kernel(const _Float16* __restrict__ xp,
                                                        const _Float16* __restrict__ wp,
                                                        const float* __restrict__ b3,
                                                        float* __restrict__ x3,
                                                        _Float16* __restrict__ x3h){
  __shared__ _Float16 Asm[2][128][40];
  __shared__ _Float16 Bs[2][32][40];
  int pt = blockIdx.x;                // 0..799  (160 rows * 5 col-tiles of 32)
  int h  = pt / 5;
  int w0 = (pt % 5) * 32;
  int cobase = blockIdx.y * 128;
  int t = threadIdx.x, lane = t & 31, wid = t >> 5;
  int l = lane >> 4;

  v8f acc0 = {};
  v8f acc1 = {};

  const _Float16* wbase = wp + (size_t)cobase*KTOT;

  if (t == 0)
    tdm_load_weights(wbase, &Asm[0][0][0]);
  conv_load_B(0, h, w0, t, xp, Bs[0]);
  if (t == 0) __builtin_amdgcn_s_wait_tensorcnt(0);
  __syncthreads();

  for (int kc=0; kc<144; ++kc){
    int cur = kc & 1;
    if (kc+1 < 144){
      int tap1   = (kc+1) >> 4;
      int cbase1 = ((kc+1) & 15) << 5;
      if (t == 0)
        tdm_load_weights(wbase + (size_t)tap1*CC + cbase1, &Asm[1-cur][0][0]);
      conv_load_B(kc+1, h, w0, t, xp, Bs[1-cur]);
    }

    // A fragment from LDS: K = e%8 + (e/8)*16 + l*8
    const _Float16* ap = &Asm[cur][wid*16 + (lane & 15)][l*8];
    v8h alo = *(const v8h*)ap;
    v8h ahi = *(const v8h*)(ap + 16);
    v16h a = __builtin_shufflevector(alo, ahi, 0,1,2,3,4,5,6,7,8,9,10,11,12,13,14,15);

    // B fragments: lane column n = pixel, K = e%8 + (e/8)*16 + l*8
    const _Float16* bp0 = &Bs[cur][lane & 15][l*8];
    const _Float16* bp1 = &Bs[cur][16 + (lane & 15)][l*8];
    v8h b0lo = *(const v8h*)bp0; v8h b0hi = *(const v8h*)(bp0 + 16);
    v8h b1lo = *(const v8h*)bp1; v8h b1hi = *(const v8h*)(bp1 + 16);
    v16h b0 = __builtin_shufflevector(b0lo, b0hi, 0,1,2,3,4,5,6,7,8,9,10,11,12,13,14,15);
    v16h b1 = __builtin_shufflevector(b1lo, b1hi, 0,1,2,3,4,5,6,7,8,9,10,11,12,13,14,15);

    acc0 = __builtin_amdgcn_wmma_f32_16x16x32_f16(false, a, false, b0, (short)0, acc0, false, false);
    acc1 = __builtin_amdgcn_wmma_f32_16x16x32_f16(false, a, false, b1, (short)0, acc1, false, false);

    if (t == 0) __builtin_amdgcn_s_wait_tensorcnt(0);  // next A chunk landed
    __syncthreads();                                   // next B stores visible
  }

  // D layout: VGPR v, lanes<16 -> M=v, lanes>=16 -> M=v+8; N = lane%16
  int co0 = cobase + wid*16 + (l << 3);
  int N = lane & 15;
  #pragma unroll
  for (int v=0; v<8; ++v){
    int co = co0 + v;
    float bias = b3[co];
    size_t base = (size_t)co*HWSZ + (size_t)h*WW_ + w0;
    float o0 = acc0[v] + bias;
    float o1 = acc1[v] + bias;
    x3[base + N]      = o0;  x3h[base + N]      = (_Float16)o0;
    x3[base + 16 + N] = o1;  x3h[base + 16 + N] = (_Float16)o1;
  }
}

// ---------------- 6) Gram GEMM: gram(512x512) = x3(512 x HW) @ x3-view(HW x 512) ----------------
__global__ __launch_bounds__(256) void gram_wmma_kernel(const _Float16* __restrict__ x3h,
                                                        float* __restrict__ gram){
  __shared__ _Float16 As[16][32];     // [row i][s within chunk]
  __shared__ _Float16 Bs[128][40];    // [col k][s within chunk], padded
  int itile    = blockIdx.x;          // 0..31
  int colbase0 = blockIdx.y * 128;    // 0..3 -> 512 cols
  int t = threadIdx.x, lane = t & 31, wid = t >> 5;
  int l = lane >> 4;
  v8f acc = {};

  for (int s0=0; s0<HWSZ; s0+=32){
    // A chunk: 16 rows x 32 s (64B per row, contiguous) via async b128 (waves 0-1)
    if (t < 64){
      int row = t >> 2, seg = t & 3;
      const _Float16* gp = x3h + (size_t)(itile*16+row)*HWSZ + s0 + seg*8;
      async_copy_b128(gp, &As[row][seg*8]);
    }
    // B chunk: 32 s x 128 cols, B[s][k] = x3_flat[s*512 + k]; store transposed
    #pragma unroll
    for (int j=0; j<16; ++j){
      int idx  = t + j*256;
      int col  = idx & 127;
      int srow = idx >> 7;
      Bs[col][srow] = x3h[(size_t)(s0+srow)*CC + colbase0 + col];
    }
    if (t < 64) wait_async0();
    __syncthreads();

    const _Float16* ap = &As[lane & 15][l*8];
    v8h alo = *(const v8h*)ap; v8h ahi = *(const v8h*)(ap + 16);
    const _Float16* bp = &Bs[wid*16 + (lane & 15)][l*8];
    v8h blo = *(const v8h*)bp; v8h bhi = *(const v8h*)(bp + 16);
    v16h a = __builtin_shufflevector(alo, ahi, 0,1,2,3,4,5,6,7,8,9,10,11,12,13,14,15);
    v16h b = __builtin_shufflevector(blo, bhi, 0,1,2,3,4,5,6,7,8,9,10,11,12,13,14,15);
    acc = __builtin_amdgcn_wmma_f32_16x16x32_f16(false, a, false, b, (short)0, acc, false, false);
    __syncthreads();
  }

  int M0 = l << 3, N = lane & 15;
  #pragma unroll
  for (int v=0; v<8; ++v)
    gram[(size_t)(itile*16 + M0 + v)*CC + colbase0 + wid*16 + N] = acc[v];
}

// ---------------- 7) per-row GroupNorm-style normalize ----------------
__global__ __launch_bounds__(256) void gn_kernel(const float* __restrict__ gram,
                                                 const float* __restrict__ gw,
                                                 const float* __restrict__ gb,
                                                 float* __restrict__ gnorm){
  __shared__ float sb[256];
  int i = blockIdx.x, t = threadIdx.x;
  float a = gram[(size_t)i*CC + t];
  float b = gram[(size_t)i*CC + t + 256];
  float s  = block_reduce_sum_256(a + b, sb);
  float sq = block_reduce_sum_256(a*a + b*b, sb);
  float mu  = s * (1.0f/CC);
  float var = sq * (1.0f/CC) - mu*mu;
  float rs  = rsqrtf(var + 1e-5f);
  gnorm[(size_t)i*CC + t]       = (a - mu)*rs*gw[t]     + gb[t];
  gnorm[(size_t)i*CC + t + 256] = (b - mu)*rs*gw[t+256] + gb[t+256];
}

// ---------------- 8) column sums of gnorm (sum over axis=1) ----------------
__global__ __launch_bounds__(256) void colsum_kernel(const float* __restrict__ gnorm,
                                                     float* __restrict__ csum){
  int t = threadIdx.x;
  #pragma unroll
  for (int rep=0; rep<2; ++rep){
    int k = t + rep*256;
    float s = 0.f;
    for (int i=0; i<CC; ++i) s += gnorm[(size_t)i*CC + k];
    csum[k] = s;
  }
}

// ---------------- 9) per-pixel channel dots + sigmoid ----------------
__global__ __launch_bounds__(256) void weights_kernel(const float* __restrict__ x21,
                                                      const float* __restrict__ x31,
                                                      const float* __restrict__ x3,
                                                      const float* __restrict__ x2,
                                                      float* __restrict__ sw){
  int n = blockIdx.x*256 + threadIdx.x;
  if (n >= HWSZ) return;
  float a = 0.f;
  for (int c=0; c<CC; ++c)
    a += x21[c]*x3[(size_t)c*HWSZ + n] + x31[c]*x2[(size_t)c*HWSZ + n];
  sw[n] = sigmoidf(a);
}

// ---------------- 10) final gate ----------------
__global__ __launch_bounds__(256) void out_kernel(const float* __restrict__ x,
                                                  const float* __restrict__ sw,
                                                  float* __restrict__ out){
  int n = blockIdx.x*256 + threadIdx.x;
  int c = blockIdx.y;
  out[(size_t)c*HWSZ + n] = x[(size_t)c*HWSZ + n] * sw[n];
}

extern "C" void kernel_launch(void* const* d_in, const int* in_sizes, int n_in,
                              void* d_out, int out_size, void* d_ws, size_t ws_size,
                              hipStream_t stream){
  (void)in_sizes; (void)n_in; (void)out_size; (void)ws_size;
  const float* x    = (const float*)d_in[0];
  const float* w1   = (const float*)d_in[1];
  const float* b1   = (const float*)d_in[2];
  const float* w3   = (const float*)d_in[3];
  const float* b3   = (const float*)d_in[4];
  const float* gn_w = (const float*)d_in[5];
  const float* gn_b = (const float*)d_in[6];
  float* out = (float*)d_out;

  char* ws = (char*)d_ws;
  size_t off = 0;
  auto alloc = [&](size_t bytes) -> void* {
    void* p = ws + off;
    off = (off + bytes + 255) & ~(size_t)255;
    return p;
  };
  float*    cat   = (float*)   alloc((size_t)CC*SLEN*sizeof(float));
  float*    sig   = (float*)   alloc((size_t)CC*SLEN*sizeof(float));
  float*    x2b   = (float*)   alloc((size_t)CC*HWSZ*sizeof(float));
  _Float16* xp    = (_Float16*)alloc((size_t)CC*PHW*sizeof(_Float16));
  _Float16* wp    = (_Float16*)alloc((size_t)CC*KTOT*sizeof(_Float16));
  float*    x3b   = (float*)   alloc((size_t)CC*HWSZ*sizeof(float));
  _Float16* x3h   = (_Float16*)alloc((size_t)CC*HWSZ*sizeof(_Float16));
  float*    mean2 = (float*)   alloc(CC*sizeof(float));
  float*    x21   = (float*)   alloc(CC*sizeof(float));
  float*    gram  = (float*)   alloc((size_t)CC*CC*sizeof(float));
  float*    gnorm = (float*)   alloc((size_t)CC*CC*sizeof(float));
  float*    csum  = (float*)   alloc(CC*sizeof(float));
  float*    x31   = (float*)   alloc(CC*sizeof(float));
  float*    sw    = (float*)   alloc((size_t)HWSZ*sizeof(float));

  pool_kernel      <<<CC, 256, 0, stream>>>(x, cat);
  hwfeat_kernel    <<<CC, 256, 0, stream>>>(w1, b1, cat, sig);
  x2_kernel        <<<CC, 256, 0, stream>>>(x, sig, x2b, xp, mean2);
  repack_w3_kernel <<<(CC*KTOT)/256, 256, 0, stream>>>(w3, wp);
  softmax512_kernel<<<1, 256, 0, stream>>>(mean2, x21);
  conv_wmma_kernel <<<dim3(800, 4), 256, 0, stream>>>(xp, wp, b3, x3b, x3h);
  gram_wmma_kernel <<<dim3(32, 4), 256, 0, stream>>>(x3h, gram);
  gn_kernel        <<<CC, 256, 0, stream>>>(gram, gn_w, gn_b, gnorm);
  colsum_kernel    <<<1, 256, 0, stream>>>(gnorm, csum);
  softmax512_kernel<<<1, 256, 0, stream>>>(csum, x31);
  weights_kernel   <<<HWSZ/256, 256, 0, stream>>>(x21, x31, x3b, x2b, sw);
  out_kernel       <<<dim3(HWSZ/256, CC), 256, 0, stream>>>(x, sw, out);
}
